// GPT_86981677678621
// MI455X (gfx1250) — compile-verified, matrix operand order checked by reading, not run on hardware
//
#include <hip/hip_runtime.h>
#include <hip/hip_bf16.h>
#include <math.h>

// ---------------- problem constants (match reference) ----------------
#define BB 2
#define KK 48
#define NP 25
#define CC 768
#define HH 12
#define LL 6
#define COLORS 10
#define VOCAB 40
#define ISZ 26
#define TOK (NP + 4)        // 29 tokens per timestep
#define TT (KK * TOK)       // 1392 sequence length
#define TP 1408             // TT padded up to a multiple of 32/64
#define HD (CC / HH)        // 64 head dim
#define MROWS (BB * TT)     // 2784 total rows

typedef __attribute__((ext_vector_type(16))) __bf16 v16bf;
typedef __attribute__((ext_vector_type(8)))  __bf16 v8bf;
typedef __attribute__((ext_vector_type(8)))  float  v8f;

__device__ __forceinline__ float gelu_exact(float x) {
  return 0.5f * x * (1.0f + erff(x * 0.70710678118654752f));
}

// ---------------------------------------------------------------------
// bf16 WMMA GEMM: Out[M,N] = act(alpha * A @ B + bias) (+resid)
//  A : bf16 row-major [M,K], lda multiple of 32, 16B-aligned rows
//  Bt: bf16 N-major  [N,K] (i.e. B transposed), ldb multiple of 32
// Per ISA 7.12.2 the 16x32 bf16 fragment per lane is two contiguous
// 8-element chunks at k = khi and k = khi+16 -> two b128 loads each.
// One wave computes a 64x64 tile (4 m-tiles x 4 n-tiles, 16 f32 accs).
// Register budget: 128 acc + 32 A-frag + 8 B-frag; addresses are 32-bit
// element offsets off uniform bases (saddr form) to avoid 64-bit VGPR
// pointer pairs. B-frags are loaded inside the i-loop and consumed
// immediately to keep liveness low (round-3 version spilled).
// Invalid m-tiles (M%64!=0) are clamped for loads (EXEC must stay all-1s
// for WMMA) and skipped in the epilogue with a wave-uniform guard.
// OUTMODE: 0 = f32 row-major, 1 = bf16 row-major, 2 = bf16 V-transpose
// Requires: M%16==0, N%64==0, K%32==0.
// ---------------------------------------------------------------------
template<int OUTMODE, int ACT>
__global__ void __launch_bounds__(256, 1)
gemm_bf16(const __bf16* __restrict__ A, int lda,
          const __bf16* __restrict__ Bt, int ldb,
          const float* __restrict__ bias,
          const float* __restrict__ resid,
          void* __restrict__ OutP, int ldo,
          int M, int N, int Kd, float alpha) {
  const int lane = threadIdx.x & 31;
  const int wave = threadIdx.x >> 5;
  const int tn0  = (blockIdx.x * 8 + wave) * 4;   // first of 4 n-tiles (uniform)
  const int tm0  = blockIdx.y * 4;                // first of 4 m-tiles (uniform)
  if (tn0 * 16 >= N) return;                      // whole-wave uniform exit

  const int khi  = (lane >= 16) ? 8 : 0;
  const int ncol = (lane & 15);

  int aoff[4];                                    // 32-bit element offsets
#pragma unroll
  for (int j = 0; j < 4; ++j) {
    const int tmj = ((tm0 + j) * 16 < M) ? (tm0 + j) : 0;   // clamp invalid tile
    aoff[j] = (tmj * 16 + (lane & 15)) * lda + khi;
  }
  const int boff = (tn0 * 16 + ncol) * ldb + khi;

  v8f acc[4][4] = {};
  for (int k0 = 0; k0 < Kd; k0 += 32) {
    v16bf af[4];
#pragma unroll
    for (int j = 0; j < 4; ++j) {
      v8bf a0 = *(const v8bf*)(A + aoff[j] + k0);
      v8bf a1 = *(const v8bf*)(A + aoff[j] + k0 + 16);
      af[j] = __builtin_shufflevector(a0, a1, 0,1,2,3,4,5,6,7,8,9,10,11,12,13,14,15);
    }
#pragma unroll
    for (int i = 0; i < 4; ++i) {
      const int bo = boff + i * 16 * ldb + k0;
      v8bf b0 = *(const v8bf*)(Bt + bo);
      v8bf b1 = *(const v8bf*)(Bt + bo + 16);
      v16bf bf = __builtin_shufflevector(b0, b1, 0,1,2,3,4,5,6,7,8,9,10,11,12,13,14,15);
#pragma unroll
      for (int j = 0; j < 4; ++j)
        acc[j][i] = __builtin_amdgcn_wmma_f32_16x16x32_bf16(false, af[j], false, bf,
                                                            (short)0, acc[j][i], false, false);
    }
  }

  // D layout: VGPR r -> M = r (lanes 0-15) / r+8 (lanes 16-31); N = lane&15
  const int rhi = (lane >= 16) ? 8 : 0;
#pragma unroll
  for (int j = 0; j < 4; ++j) {
    if ((tm0 + j) * 16 < M) {                    // wave-uniform store guard
      const int mbase = (tm0 + j) * 16 + rhi;
#pragma unroll
      for (int i = 0; i < 4; ++i) {
        const int n = (tn0 + i) * 16 + ncol;
        const float bv = bias ? bias[n] : 0.0f;
#pragma unroll
        for (int r = 0; r < 8; ++r) {
          const int mo = mbase + r;
          float v = acc[j][i][r] * alpha + bv;
          if (ACT == 1) v = gelu_exact(v);
          if (resid) v += resid[(size_t)mo * ldo + n];
          if (OUTMODE == 0) {
            ((float*)OutP)[(size_t)mo * ldo + n] = v;
          } else if (OUTMODE == 1) {
            ((__bf16*)OutP)[(size_t)mo * ldo + n] = (__bf16)v;
          } else {  // V-transpose store: Vt[((b*H + h)*HD + d) * TP + t]
            const int b = mo / TT, t = mo % TT;
            const int h = n / HD,  d = n % HD;
            ((__bf16*)OutP)[((size_t)(b * HH + h) * HD + d) * TP + t] = (__bf16)v;
          }
        }
      }
    }
  }
}

// --------- weight convert+transpose: W[l][k][n] f32 -> Out[l][n][k] bf16 ----
__global__ void conv_transpose_k(const float* __restrict__ W, __bf16* __restrict__ Out,
                                 int Kd, int N) {
  const size_t base = (size_t)blockIdx.z * Kd * N;
  const size_t tot  = (size_t)Kd * N;
  for (size_t idx = blockIdx.x * 256 + threadIdx.x; idx < tot; idx += (size_t)gridDim.x * 256) {
    const int n = (int)(idx / Kd), k = (int)(idx % Kd);     // writes coalesced in k
    Out[base + idx] = (__bf16)W[base + (size_t)k * N + n];
  }
}

// --------- zero the padded tail columns of Vt (rows B*C, 16 cols each) -----
__global__ void vt_tail_zero_k(__bf16* __restrict__ Vt) {
  const int idx = blockIdx.x * 256 + threadIdx.x;
  if (idx >= BB * CC * (TP - TT)) return;
  const int row = idx / (TP - TT), j = idx % (TP - TT);
  Vt[(size_t)row * TP + TT + j] = (__bf16)0.0f;
}

// ---------------- LayerNorm: one 256-thread block per row ------------------
template<typename OT>
__global__ void layernorm_k(const float* __restrict__ in,
                            const float* __restrict__ g,
                            const float* __restrict__ b,
                            OT* __restrict__ out) {
  const int row = blockIdx.x;
  const float* x = in + (size_t)row * CC;
  __shared__ float red[256];
  float s = 0.0f;
  for (int c = threadIdx.x; c < CC; c += 256) s += x[c];
  red[threadIdx.x] = s; __syncthreads();
  for (int o = 128; o > 0; o >>= 1) {
    if (threadIdx.x < o) red[threadIdx.x] += red[threadIdx.x + o];
    __syncthreads();
  }
  const float mean = red[0] * (1.0f / CC);
  __syncthreads();
  float v = 0.0f;
  for (int c = threadIdx.x; c < CC; c += 256) { float d = x[c] - mean; v += d * d; }
  red[threadIdx.x] = v; __syncthreads();
  for (int o = 128; o > 0; o >>= 1) {
    if (threadIdx.x < o) red[threadIdx.x] += red[threadIdx.x + o];
    __syncthreads();
  }
  const float rstd = rsqrtf(red[0] * (1.0f / CC) + 1e-5f);
  for (int c = threadIdx.x; c < CC; c += 256)
    out[(size_t)row * CC + c] = (OT)((x[c] - mean) * rstd * g[c] + b[c]);
}

// ------ causal softmax: f32 scores row (ld TP) -> bf16 probs row (ld TP) ---
__global__ void softmax_causal_k(float* __restrict__ in, __bf16* __restrict__ out) {
  const int t = blockIdx.x;
  float* row = in + (size_t)t * TP;
  __bf16* orow = out + (size_t)t * TP;
  const int n = t + 1;                 // valid columns 0..t
  __shared__ float red[256];
  float m = -3.0e38f;
  for (int j = threadIdx.x; j < n; j += 256) m = fmaxf(m, row[j]);
  red[threadIdx.x] = m; __syncthreads();
  for (int o = 128; o > 0; o >>= 1) {
    if (threadIdx.x < o) red[threadIdx.x] = fmaxf(red[threadIdx.x], red[threadIdx.x + o]);
    __syncthreads();
  }
  m = red[0]; __syncthreads();
  float s = 0.0f;
  for (int j = threadIdx.x; j < n; j += 256) {
    float e = __expf(row[j] - m); row[j] = e; s += e;
  }
  red[threadIdx.x] = s; __syncthreads();
  for (int o = 128; o > 0; o >>= 1) {
    if (threadIdx.x < o) red[threadIdx.x] += red[threadIdx.x + o];
    __syncthreads();
  }
  const float inv = 1.0f / red[0];
  for (int j = threadIdx.x; j < TP; j += 256)
    orow[j] = (__bf16)((j < n) ? row[j] * inv : 0.0f);
}

// ------------- embedding + positional encodings, one block per (b,k) -------
__global__ void embed_k(const int* __restrict__ states, const int* __restrict__ actions,
                        const float* __restrict__ rtgs, const int* __restrict__ timesteps,
                        const float* __restrict__ pnp, const int* __restrict__ intentions,
                        const float* __restrict__ pos_emb, const float* __restrict__ gpe,
                        const float* __restrict__ state_tab, const float* __restrict__ action_tab,
                        const float* __restrict__ ret_W, const float* __restrict__ ret_b,
                        const float* __restrict__ pnp_W, const float* __restrict__ pnp_b,
                        const float* __restrict__ src_tab, const float* __restrict__ dst_tab,
                        const float* __restrict__ int_W, const float* __restrict__ int_b,
                        float* __restrict__ X) {
  const int bk = blockIdx.x;
  const int b = bk / KK, k = bk % KK;
  const int ts  = timesteps[bk];
  const int act = actions[bk];
  const float rtg = rtgs[bk];
  const int i0 = intentions[bk * 2 + 0];
  const int i1 = intentions[bk * 2 + 1];
  const int*   srow = states + bk * NP;
  const float* prow = pnp + bk * NP;
  const size_t rowbase = (size_t)(b * TT + k * TOK) * CC;
  for (int c = threadIdx.x; c < CC; c += 256) {
    const float pos = gpe[(size_t)ts * CC + c] + pos_emb[(size_t)k * CC + c];
    X[rowbase + c] = rtg * ret_W[c] + ret_b[c] + pos;                      // rtg token
    for (int i = 0; i < NP; ++i)                                           // 25 states
      X[rowbase + (size_t)(1 + i) * CC + c] = state_tab[(size_t)srow[i] * CC + c] + pos;
    X[rowbase + (size_t)(NP + 1) * CC + c] = action_tab[(size_t)act * CC + c] + pos;
    float pv = pnp_b[c];                                                   // pnp proj
    for (int i = 0; i < NP; ++i) pv += prow[i] * pnp_W[(size_t)i * CC + c];
    X[rowbase + (size_t)(NP + 2) * CC + c] = pv + pos;
    float iv = int_b[c];                                                   // intention
    for (int j = 0; j < CC; ++j) {
      iv += src_tab[(size_t)i0 * CC + j] * int_W[(size_t)j * CC + c];
      iv += dst_tab[(size_t)i1 * CC + j] * int_W[(size_t)(CC + j) * CC + c];
    }
    X[rowbase + (size_t)(NP + 3) * CC + c] = iv + pos;
  }
}

// ------------- output heads (tiny N: scalar dot products) ------------------
__global__ void head_all_k(const float* __restrict__ ln, const float* __restrict__ W,
                           const float* __restrict__ bias, float* __restrict__ out) {
  const int idx = blockIdx.x * blockDim.x + threadIdx.x;
  if (idx >= MROWS * COLORS) return;
  const int row = idx / COLORS, col = idx % COLORS;
  float s = bias[col];
  for (int j = 0; j < CC; ++j) s += ln[(size_t)row * CC + j] * W[(size_t)j * COLORS + col];
  out[idx] = s;
}

__global__ void head_tok_k(const float* __restrict__ ln, const float* __restrict__ W,
                           const float* __restrict__ bias,
                           float* __restrict__ out1, float* __restrict__ out2,
                           int N, int split, int tokoff) {
  const int idx = blockIdx.x * blockDim.x + threadIdx.x;
  if (idx >= BB * KK * N) return;
  const int bk = idx / N, col = idx % N;
  const int b = bk / KK, k = bk % KK;
  const int row = b * TT + k * TOK + tokoff;
  float s = bias[col];
  for (int j = 0; j < CC; ++j) s += ln[(size_t)row * CC + j] * W[(size_t)j * N + col];
  if (col < split) out1[bk * split + col] = s;
  else             out2[bk * (N - split) + (col - split)] = s;
}

// ---------------------------------------------------------------------------
static void launch_gemm(hipStream_t st, const __bf16* A, int lda,
                        const __bf16* Bt, int ldb, const float* bias,
                        const float* resid, void* Out, int ldo,
                        int M, int N, int Kd, float alpha, int outmode, int act) {
  dim3 blk(256);
  dim3 grd((unsigned)((N + 511) / 512), (unsigned)((M / 16 + 3) / 4));
  if (outmode == 0)
    gemm_bf16<0, 0><<<grd, blk, 0, st>>>(A, lda, Bt, ldb, bias, resid, Out, ldo, M, N, Kd, alpha);
  else if (outmode == 2)
    gemm_bf16<2, 0><<<grd, blk, 0, st>>>(A, lda, Bt, ldb, bias, resid, Out, ldo, M, N, Kd, alpha);
  else if (act == 1)
    gemm_bf16<1, 1><<<grd, blk, 0, st>>>(A, lda, Bt, ldb, bias, resid, Out, ldo, M, N, Kd, alpha);
  else
    gemm_bf16<1, 0><<<grd, blk, 0, st>>>(A, lda, Bt, ldb, bias, resid, Out, ldo, M, N, Kd, alpha);
}

static inline size_t align256(size_t x) { return (x + 255) & ~(size_t)255; }

extern "C" void kernel_launch(void* const* d_in, const int* in_sizes, int n_in,
                              void* d_out, int out_size, void* d_ws, size_t ws_size,
                              hipStream_t stream) {
  (void)in_sizes; (void)n_in; (void)out_size; (void)ws_size;
  const int*   states     = (const int*)  d_in[0];
  const int*   actions    = (const int*)  d_in[1];
  const float* rtgs       = (const float*)d_in[2];
  const int*   timesteps  = (const int*)  d_in[3];
  const float* pnp        = (const float*)d_in[4];
  const int*   intentions = (const int*)  d_in[5];
  const float* pos_emb    = (const float*)d_in[6];
  const float* gpe        = (const float*)d_in[7];
  const float* state_tab  = (const float*)d_in[8];
  const float* action_tab = (const float*)d_in[9];
  const float* ret_W      = (const float*)d_in[10];
  const float* ret_b      = (const float*)d_in[11];
  const float* pnp_W      = (const float*)d_in[12];
  const float* pnp_b      = (const float*)d_in[13];
  const float* src_tab    = (const float*)d_in[14];
  const float* dst_tab    = (const float*)d_in[15];
  const float* int_W      = (const float*)d_in[16];
  const float* int_b      = (const float*)d_in[17];
  const float* ln1g       = (const float*)d_in[18];
  const float* ln1b       = (const float*)d_in[19];
  const float* ln2g       = (const float*)d_in[20];
  const float* ln2b       = (const float*)d_in[21];
  const float* Wq         = (const float*)d_in[22];
  const float* bq         = (const float*)d_in[23];
  const float* Wk         = (const float*)d_in[24];
  const float* bk         = (const float*)d_in[25];
  const float* Wv         = (const float*)d_in[26];
  const float* bv         = (const float*)d_in[27];
  const float* Wo         = (const float*)d_in[28];
  const float* bo         = (const float*)d_in[29];
  const float* Wm1        = (const float*)d_in[30];
  const float* bm1        = (const float*)d_in[31];
  const float* Wm2        = (const float*)d_in[32];
  const float* bm2        = (const float*)d_in[33];
  const float* lnfg       = (const float*)d_in[34];
  const float* lnfb       = (const float*)d_in[35];
  const float* Whs        = (const float*)d_in[36];
  const float* bhs        = (const float*)d_in[37];
  const float* Wha        = (const float*)d_in[38];
  const float* bha        = (const float*)d_in[39];
  const float* Whr        = (const float*)d_in[40];
  const float* bhr        = (const float*)d_in[41];
  const float* Whi        = (const float*)d_in[42];
  const float* bhi        = (const float*)d_in[43];

  // ---------------- workspace carve-up (256B aligned partitions) ----------
  const size_t MC = (size_t)MROWS * CC;
  char* p = (char*)d_ws;
  float*  X     = (float*)p;            p += align256(MC * 4);
  __bf16* LNbf  = (__bf16*)p;           p += align256(MC * 2);     // also Ybf
  __bf16* Qbf   = (__bf16*)p;           p += align256(MC * 2);
  __bf16* Kbf   = (__bf16*)p;           p += align256(MC * 2);
  __bf16* Vt    = (__bf16*)p;           p += align256((size_t)BB * CC * TP * 2);
  // scratch region (time-shared): HIDbf | (ATTf + ATTbf) | LNf(final)
  char*   scr   = p;
  __bf16* HIDbf = (__bf16*)scr;                                    // M x 4C bf16
  float*  ATTf  = (float*)scr;                                     // T x TP f32
  __bf16* ATTbf = (__bf16*)(scr + align256((size_t)TT * TP * 4));  // T x TP bf16
  float*  LNf   = (float*)scr;                                     // M x C f32 (final)
  p = scr + align256((size_t)MROWS * 4 * CC * 2);
  // bf16 transposed weights [l][n][k]
  __bf16* Wqt  = (__bf16*)p;  p += align256((size_t)LL * CC * CC * 2);
  __bf16* Wkt  = (__bf16*)p;  p += align256((size_t)LL * CC * CC * 2);
  __bf16* Wvt  = (__bf16*)p;  p += align256((size_t)LL * CC * CC * 2);
  __bf16* Wot  = (__bf16*)p;  p += align256((size_t)LL * CC * CC * 2);
  __bf16* W1t  = (__bf16*)p;  p += align256((size_t)LL * CC * 4 * CC * 2);
  __bf16* W2t  = (__bf16*)p;  p += align256((size_t)LL * CC * 4 * CC * 2);

  // 0) one-time weight convert+transpose to N-major bf16, zero Vt pad tails
  conv_transpose_k<<<dim3(512, 1, LL), 256, 0, stream>>>(Wq, Wqt, CC, CC);
  conv_transpose_k<<<dim3(512, 1, LL), 256, 0, stream>>>(Wk, Wkt, CC, CC);
  conv_transpose_k<<<dim3(512, 1, LL), 256, 0, stream>>>(Wv, Wvt, CC, CC);
  conv_transpose_k<<<dim3(512, 1, LL), 256, 0, stream>>>(Wo, Wot, CC, CC);
  conv_transpose_k<<<dim3(1024, 1, LL), 256, 0, stream>>>(Wm1, W1t, CC, 4 * CC);
  conv_transpose_k<<<dim3(1024, 1, LL), 256, 0, stream>>>(Wm2, W2t, 4 * CC, CC);
  vt_tail_zero_k<<<(BB * CC * (TP - TT) + 255) / 256, 256, 0, stream>>>(Vt);

  // 1) embeddings + positions (f32 residual stream X)
  embed_k<<<BB * KK, 256, 0, stream>>>(states, actions, rtgs, timesteps, pnp,
                                       intentions, pos_emb, gpe, state_tab,
                                       action_tab, ret_W, ret_b, pnp_W, pnp_b,
                                       src_tab, dst_tab, int_W, int_b, X);

  // 2) transformer stack
  for (int l = 0; l < LL; ++l) {
    const __bf16* wqt = Wqt + (size_t)l * CC * CC;
    const __bf16* wkt = Wkt + (size_t)l * CC * CC;
    const __bf16* wvt = Wvt + (size_t)l * CC * CC;
    const __bf16* wot = Wot + (size_t)l * CC * CC;
    const __bf16* w1t = W1t + (size_t)l * CC * 4 * CC;
    const __bf16* w2t = W2t + (size_t)l * CC * 4 * CC;

    layernorm_k<__bf16><<<MROWS, 256, 0, stream>>>(X, ln1g + l * CC, ln1b + l * CC, LNbf);
    launch_gemm(stream, LNbf, CC, wqt, CC, bq + l * CC, nullptr, Qbf, CC, MROWS, CC, CC, 1.0f, 1, 0);
    launch_gemm(stream, LNbf, CC, wkt, CC, bk + l * CC, nullptr, Kbf, CC, MROWS, CC, CC, 1.0f, 1, 0);
    launch_gemm(stream, LNbf, CC, wvt, CC, bv + l * CC, nullptr, Vt,  0,  MROWS, CC, CC, 1.0f, 2, 0);

    for (int b = 0; b < BB; ++b)
      for (int h = 0; h < HH; ++h) {
        const __bf16* qh  = Qbf + (size_t)b * TT * CC + h * HD;
        const __bf16* kh  = Kbf + (size_t)b * TT * CC + h * HD;   // N-major for k^T
        const __bf16* vth = Vt  + (size_t)(b * HH + h) * HD * TP; // [HD][TP]
        __bf16*       yh  = LNbf + (size_t)b * TT * CC + h * HD;  // Y aliases LNbf
        // scores = (q @ k^T) / sqrt(hd); pad columns 1392..1407 are masked later
        launch_gemm(stream, qh, CC, kh, CC, nullptr, nullptr, ATTf, TP, TT, TP, HD, 0.125f, 0, 0);
        softmax_causal_k<<<TT, 256, 0, stream>>>(ATTf, ATTbf);
        // y = probs @ v  (K = TP, pad probs/V are zero)
        launch_gemm(stream, ATTbf, TP, vth, TP, nullptr, nullptr, yh, CC, TT, HD, TP, 1.0f, 1, 0);
      }

    // x = x + Y @ Wo + bo
    launch_gemm(stream, LNbf, CC, wot, CC, bo + l * CC, X, X, CC, MROWS, CC, CC, 1.0f, 0, 0);
    // MLP
    layernorm_k<__bf16><<<MROWS, 256, 0, stream>>>(X, ln2g + l * CC, ln2b + l * CC, LNbf);
    launch_gemm(stream, LNbf, CC, w1t, CC, bm1 + (size_t)l * 4 * CC, nullptr,
                HIDbf, 4 * CC, MROWS, 4 * CC, CC, 1.0f, 1, 1);
    launch_gemm(stream, HIDbf, 4 * CC, w2t, 4 * CC, bm2 + l * CC, X,
                X, CC, MROWS, CC, 4 * CC, 1.0f, 0, 0);
  }

  // 3) final LN (f32 for heads) + heads
  layernorm_k<float><<<MROWS, 256, 0, stream>>>(X, lnfg, lnfb, LNf);
  float* out = (float*)d_out;
  float* o0 = out;                                // logits     (B,T,COLORS)
  float* o1 = o0 + (size_t)MROWS * COLORS;        // action     (B,K,VOCAB)
  float* o2 = o1 + (size_t)BB * KK * VOCAB;       // rtg        (B,K)
  float* o3 = o2 + (size_t)BB * KK;               // int src    (B,K,IS)
  float* o4 = o3 + (size_t)BB * KK * ISZ;         // int dst    (B,K,IS)
  head_all_k<<<(MROWS * COLORS + 255) / 256, 256, 0, stream>>>(LNf, Whs, bhs, o0);
  head_tok_k<<<(BB * KK * VOCAB + 255) / 256, 256, 0, stream>>>(LNf, Wha, bha, o1, o1, VOCAB, VOCAB, NP + 1);
  head_tok_k<<<(BB * KK + 255) / 256, 256, 0, stream>>>(LNf, Whr, bhr, o2, o2, 1, 1, 0);
  head_tok_k<<<(BB * KK * 2 * ISZ + 255) / 256, 256, 0, stream>>>(LNf, Whi, bhi, o3, o4, 2 * ISZ, ISZ, NP + 3);
}
